// Feature2VertexLayer_79147657331110
// MI455X (gfx1250) — matrix operand order repaired and checked by reading, not compiled, not run on hardware
//
#include <hip/hip_runtime.h>

typedef __attribute__((ext_vector_type(16))) __bf16       v16bf;
typedef __attribute__((ext_vector_type(8)))  float        v8f;
typedef __attribute__((ext_vector_type(4)))  unsigned int v4u;
typedef __attribute__((ext_vector_type(8)))  unsigned int v8u;

#define N_NODES 200000
#define N_EDGES 600000
#define MTILES  (N_NODES / 16)   // 12500 exact tiles
#define WPB     8                // waves per block (256 threads, wave32)

__device__ __forceinline__ unsigned short f2bf(float f) {
  unsigned int u = __float_as_uint(f);
  u += 0x7FFFu + ((u >> 16) & 1u);          // round-to-nearest-even
  return (unsigned short)(u >> 16);
}

union BfCast { v8u u; v16bf b; };

__global__ void convert_f32_bf16(const float* __restrict__ in,
                                 unsigned short* __restrict__ out, long long n) {
  long long i = (long long)blockIdx.x * blockDim.x + threadIdx.x;
  if (i < n) out[i] = f2bf(in[i]);
}

__global__ void relu_f32_bf16(const float* __restrict__ in,
                              unsigned short* __restrict__ out, long long n) {
  long long i = (long long)blockIdx.x * blockDim.x + threadIdx.x;
  if (i < n) { float v = in[i]; out[i] = f2bf(v > 0.0f ? v : 0.0f); }
}

// wT[n][k] = w[k][n] as bf16, zero-padded for n >= dout (n < dpad)
__global__ void prep_weights(const float* __restrict__ w,
                             unsigned short* __restrict__ wT,
                             int din, int dout, int dpad) {
  int i = blockIdx.x * blockDim.x + threadIdx.x;
  if (i >= dpad * din) return;
  int n = i / din, k = i - n * din;
  float v = (n < dout) ? w[(size_t)k * dout + n] : 0.0f;
  wT[i] = f2bf(v);
}

// One wave computes one 16(M)x16(N) tile of BOTH h0 = x@w0+b0 and h1 = x@w1+b1.
// A (bf16 16x32) layout: lane<16 holds row (lane&15), K = kk+h*16+{0..15} (h=lane>>4, contiguous).
// B (bf16 32x16) layout: lane holds column (lane&15), K = kk+h*16+{0..15} (contiguous).
__global__ __launch_bounds__(256) void gconv_gemm(
    const unsigned short* __restrict__ xb,
    const unsigned short* __restrict__ wT0,
    const unsigned short* __restrict__ wT1,
    const float* __restrict__ b0, const float* __restrict__ b1,
    float* __restrict__ h0, float* __restrict__ h1,
    int din, int dout) {
  const int wid   = threadIdx.x >> 5;
  const int lane  = threadIdx.x & 31;
  const int mtile = blockIdx.x * WPB + wid;
  if (mtile >= MTILES) return;                    // wave-uniform guard

  const int half = lane >> 4;
  const int nl   = lane & 15;
  const int ng   = blockIdx.y * 16 + nl;          // column in padded N space
  const bool nok = (ng < dout);
  const float bias0 = nok ? b0[ng] : 0.0f;
  const float bias1 = nok ? b1[ng] : 0.0f;

  v8f acc0, acc1;
#pragma unroll
  for (int v = 0; v < 8; ++v) { acc0[v] = bias0; acc1[v] = bias1; }

  const int row = mtile * 16 + nl;                // A row index for this lane
  const unsigned short* ap  = xb  + (size_t)row * din + half * 16;
  const unsigned short* bp0 = wT0 + (size_t)ng  * din + half * 16;
  const unsigned short* bp1 = wT1 + (size_t)ng  * din + half * 16;

  for (int kk = 0; kk < din; kk += 32) {
    const v4u* pa = (const v4u*)(ap + kk);
    v4u alo = pa[0];                              // K = kk+h*16 .. +7
    v4u ahi = pa[1];                              // K = kk+h*16+8 .. +15
    BfCast a;
    a.u = (v8u){alo[0], alo[1], alo[2], alo[3], ahi[0], ahi[1], ahi[2], ahi[3]};
    BfCast bb0; bb0.u = *(const v8u*)(bp0 + kk);  // 32 contiguous bytes
    BfCast bb1; bb1.u = *(const v8u*)(bp1 + kk);
    acc0 = __builtin_amdgcn_wmma_f32_16x16x32_bf16(false, a.b, false, bb0.b,
                                                   (short)0, acc0, false, false);
    acc1 = __builtin_amdgcn_wmma_f32_16x16x32_bf16(false, a.b, false, bb1.b,
                                                   (short)0, acc1, false, false);
  }

  if (nok) {
#pragma unroll
    for (int v = 0; v < 8; ++v) {                 // D: lane<16 -> M=v, lane>=16 -> M=v+8
      size_t r = (size_t)(mtile * 16 + v + half * 8) * dout + ng;
      h0[r] = acc0[v];
      h1[r] = acc1[v];
    }
  }
}

// out[src] += h1[dst]; out[dst] += h1[src]  (one thread per edge-channel)
__global__ void scatter_edges(const int* __restrict__ edges,
                              const float* __restrict__ h1,
                              float* out, int dout, long long total) {
  long long i = (long long)blockIdx.x * blockDim.x + threadIdx.x;
  if (i >= total) return;
  long long e = i / dout;
  int c = (int)(i - e * dout);
  int s = edges[2 * e], d = edges[2 * e + 1];
  atomicAdd(out + (size_t)s * dout + c, h1[(size_t)d * dout + c]);
  atomicAdd(out + (size_t)d * dout + c, h1[(size_t)s * dout + c]);
}

extern "C" void kernel_launch(void* const* d_in, const int* in_sizes, int n_in,
                              void* d_out, int out_size, void* d_ws, size_t ws_size,
                              hipStream_t stream) {
  (void)in_sizes; (void)n_in; (void)out_size; (void)ws_size;
  const float* features = (const float*)d_in[0];
  const int*   edges    = (const int*)d_in[1];
  const float *W0[4], *B0[4], *W1[4], *B1[4];
  for (int l = 0; l < 4; ++l) {
    W0[l] = (const float*)d_in[2 + 4 * l];
    B0[l] = (const float*)d_in[3 + 4 * l];
    W1[l] = (const float*)d_in[4 + 4 * l];
    B1[l] = (const float*)d_in[5 + 4 * l];
  }
  static const int DIN[4]  = {256, 192, 128, 64};
  static const int DOUT[4] = {192, 128, 64, 3};
  static const int DPAD[4] = {192, 128, 64, 16};   // N padded to WMMA tile

  char* ws = (char*)d_ws;
  size_t off = 0;
  auto alloc = [&](size_t bytes) -> char* {
    char* p = ws + off;
    off = (off + bytes + 255) & ~(size_t)255;
    return p;
  };
  unsigned short* xA   = (unsigned short*)alloc((size_t)N_NODES * 256 * 2);
  unsigned short* xB   = (unsigned short*)alloc((size_t)N_NODES * 192 * 2);
  float*          hout = (float*)alloc((size_t)N_NODES * 192 * 4);
  float*          h1b  = (float*)alloc((size_t)N_NODES * 192 * 4);
  unsigned short *WT0[4], *WT1[4];
  for (int l = 0; l < 4; ++l) {
    WT0[l] = (unsigned short*)alloc((size_t)DPAD[l] * DIN[l] * 2);
    WT1[l] = (unsigned short*)alloc((size_t)DPAD[l] * DIN[l] * 2);
  }

  { long long n = (long long)N_NODES * 256;
    convert_f32_bf16<<<(unsigned)((n + 255) / 256), 256, 0, stream>>>(features, xA, n); }

  for (int l = 0; l < 4; ++l) {
    int tot = DPAD[l] * DIN[l];
    prep_weights<<<(tot + 255) / 256, 256, 0, stream>>>(W0[l], WT0[l], DIN[l], DOUT[l], DPAD[l]);
    prep_weights<<<(tot + 255) / 256, 256, 0, stream>>>(W1[l], WT1[l], DIN[l], DOUT[l], DPAD[l]);
  }

  unsigned short* xin = xA;
  unsigned short* xnext = xB;
  for (int l = 0; l < 4; ++l) {
    float* out_f = (l == 3) ? (float*)d_out : hout;
    dim3 grid((MTILES + WPB - 1) / WPB, DPAD[l] / 16);
    gconv_gemm<<<grid, 32 * WPB, 0, stream>>>(xin, WT0[l], WT1[l], B0[l], B1[l],
                                              out_f, h1b, DIN[l], DOUT[l]);
    long long total = (long long)N_EDGES * DOUT[l];
    scatter_edges<<<(unsigned)((total + 255) / 256), 256, 0, stream>>>(
        edges, h1b, out_f, DOUT[l], total);
    if (l < 3) {
      long long n = (long long)N_NODES * DOUT[l];
      relu_f32_bf16<<<(unsigned)((n + 255) / 256), 256, 0, stream>>>(out_f, xnext, n);
      unsigned short* t = xin; xin = xnext; xnext = t;
    }
  }
}
